// LocalFusionModule_mask_9620726743503
// MI455X (gfx1250) — matrix operand order repaired and checked by reading, not compiled, not run on hardware
//
#include <hip/hip_runtime.h>
#include <hip/hip_bf16.h>

#ifndef __has_builtin
#define __has_builtin(x) 0
#endif

#if __has_builtin(__builtin_amdgcn_tensor_load_to_lds)
#define LSTM_TDM 1
#else
#define LSTM_TDM 0
#endif

// ---------------------------------------------------------------------------
// Types & helpers
// ---------------------------------------------------------------------------
typedef unsigned short     u16;
typedef unsigned int       u32;
typedef unsigned long long u64;
typedef __attribute__((ext_vector_type(16))) __bf16 v16bf;
typedef __attribute__((ext_vector_type(8)))  float  v8f;
typedef __attribute__((ext_vector_type(4)))  u32    v4u;
typedef __attribute__((ext_vector_type(8)))  int    v8i;
typedef __attribute__((ext_vector_type(4)))  int    v4i;

union FragU {
    v16bf v;
    v4u   q[2];
    u16   h[16];
};
union Pack8 {
    v4u q;
    u16 h[8];
};

__device__ __forceinline__ u16 f2bf(float f) {
    u32 u = __float_as_uint(f);
    u32 r = (u + 0x7FFFu + ((u >> 16) & 1u)) >> 16;   // round-to-nearest-even
    return (u16)r;
}
__device__ __forceinline__ float bf2f(u16 h) {
    return __uint_as_float(((u32)h) << 16);
}
__device__ __forceinline__ float sigm(float x) {
    return 1.0f / (1.0f + __expf(-x));
}
__device__ __forceinline__ float tanh_fast(float x) {
    x = fminf(fmaxf(x, -15.0f), 15.0f);
    float e = __expf(2.0f * x);
    return (e - 1.0f) / (e + 1.0f);
}

// A-matrix fragment (16x32 bf16): lane holds row M=lane&15,
// 16 contiguous K at base = lane<16 ? 0 : 16.
__device__ __forceinline__ v16bf load_a_frag(const u16* row, int kk, int lane) {
    FragU f;
    const int base = (lane < 16) ? 0 : 16;
    f.q[0] = *(const v4u*)(row + kk + base);
    f.q[1] = *(const v4u*)(row + kk + base + 8);
    return f.v;
}
// B-matrix fragment (32x16 bf16): lane holds column N=lane&15,
// 16 contiguous K at base = lane<16 ? 0 : 16.
__device__ __forceinline__ v16bf load_b_frag(const u16* row, int kk, int lane) {
    FragU f;
    const int base = (lane < 16) ? 0 : 16;
    f.q[0] = *(const v4u*)(row + kk + base);
    f.q[1] = *(const v4u*)(row + kk + base + 8);
    return f.v;
}
__device__ __forceinline__ v8f wmma_bf16(v16bf a, v16bf b, v8f c) {
    return __builtin_amdgcn_wmma_f32_16x16x32_bf16(
        /*neg_a=*/false, a, /*neg_b=*/false, b,
        /*c_mod=*/(short)0, c, /*reuse_a=*/false, /*reuse_b=*/false);
}

#define BB    32
#define NT    9
#define NREF  8
#define CC    128
#define HW    576
#define GID   512          // 4*HID
#define STEPG (BB * GID)   // 16384 bf16 elements = 32 KB per timestep

#if LSTM_TDM
// TDM: contiguous 32KB (16384 x 2B) tile from global -> LDS, per CDNA5 ISA.
// D# group0: count=1, lds_addr, global_addr[56:0], type=2.
// D# group1: data_size=2B; tensor_dim0=tile_dim0=16384; tensor_dim1=tile_dim1=1;
//            tensor_dim0_stride=16384.
__device__ __forceinline__ void tdm_load_gin(const u16* gsrc, u32 lds_off) {
    u64 ga = (u64)(size_t)gsrc;
    v4u g0;
    g0[0] = 1u;                                            // count=1 (user mode)
    g0[1] = lds_off;                                       // lds_addr (bytes)
    g0[2] = (u32)ga;                                       // global_addr[31:0]
    g0[3] = ((u32)(ga >> 32) & 0x01FFFFFFu) | (2u << 30);  // addr[56:32] | type=2
    v8i g1;
    g1[0] = 0x00010000;   // workgroup_mask=0, data_size=1 (2 bytes)
    g1[1] = 0x40000000;   // tensor_dim0 = 16384 (low16 in bits[31:16])
    g1[2] = 0x00010000;   // tensor_dim0 hi=0, tensor_dim1 = 1
    g1[3] = 0x40000000;   // tensor_dim1 hi=0, tile_dim0 = 16384
    g1[4] = 1;            // tile_dim1 = 1, tile_dim2 = 0
    g1[5] = 16384;        // tensor_dim0_stride low32
    g1[6] = 0;            // stride0 hi, tensor_dim1_stride low
    g1[7] = 0;
    v4i gz = {0, 0, 0, 0};
#if __clang_major__ >= 23
    v8i gz8 = {0, 0, 0, 0, 0, 0, 0, 0};
    __builtin_amdgcn_tensor_load_to_lds(g0, g1, gz, gz, gz8, 0);
#else
    __builtin_amdgcn_tensor_load_to_lds(g0, g1, gz, gz, 0);
#endif
}
#endif

// ---------------------------------------------------------------------------
// Kernel 1: mask -> indices (tile nonzero positions to length q)
// ---------------------------------------------------------------------------
__global__ void k_indices(const float* __restrict__ mask, const int* __restrict__ indexp,
                          int* __restrict__ idxs, int* __restrict__ tmp,
                          float* __restrict__ out_idx, int q) {
    int b = blockIdx.x;
    if (threadIdx.x != 0) return;
    int index = *indexp;
    const float* m = mask + (size_t)(b * NT + index) * HW;
    int* t = tmp + b * HW;
    int k = 0;
    for (int p = 0; p < HW; ++p)
        if (m[p] > 0.0f) t[k++] = p;
    if (k == 0) { t[0] = 0; k = 1; }
    for (int i = 0; i < q; ++i) {
        int v = t[i % k];
        idxs[b * q + i] = v;
        out_idx[b * q + i] = (float)v;
    }
}

// ---------------------------------------------------------------------------
// Kernel 2: gather + L2-normalize selected feat columns -> wfs (b,q,128) bf16
// ---------------------------------------------------------------------------
__global__ void k_wfs(const float* __restrict__ feat, const int* __restrict__ idxs,
                      u16* __restrict__ wfs, int q) {
    int qi = blockIdx.x, b = blockIdx.y, c = threadIdx.x;
    int p = idxs[b * q + qi];
    float v = feat[((size_t)(b * CC + c)) * HW + p];
    __shared__ float red[CC];
    red[c] = v * v;
    __syncthreads();
    for (int s = CC / 2; s > 0; s >>= 1) {
        if (c < s) red[c] += red[c + s];
        __syncthreads();
    }
    float inv = 1.0f / fmaxf(sqrtf(red[0]), 1e-12f);
    wfs[((size_t)(b * q + qi)) * CC + c] = f2bf(v * inv);
}

// ---------------------------------------------------------------------------
// Kernel 3: L2-normalize refs over c and transpose -> wrt (b,8,576,128) bf16
// ---------------------------------------------------------------------------
__global__ void k_wrt(const float* __restrict__ refs, const int* __restrict__ indexp,
                      u16* __restrict__ wrt) {
    int t = blockIdx.x;                  // b*8*576 blocks
    int c = threadIdx.x;
    int p = t % HW;
    int s = (t / HW) % NREF;
    int b = t / (HW * NREF);
    int index = *indexp;
    int n0 = (s < index) ? s : s + 1;
    float v = refs[(((size_t)(b * NT + n0) * CC) + c) * HW + p];
    __shared__ float red[CC];
    red[c] = v * v;
    __syncthreads();
    for (int st = CC / 2; st > 0; st >>= 1) {
        if (c < st) red[c] += red[c + st];
        __syncthreads();
    }
    float inv = 1.0f / fmaxf(sqrtf(red[0]), 1e-12f);
    wrt[((size_t)(b * NREF + s) * HW + p) * CC + c] = f2bf(v * inv);
}

// ---------------------------------------------------------------------------
// Kernel 4: fx = wfs(q,c) x wrt(c,h) via WMMA bf16, fused argmax over h
// grid (qTiles, 8, B), 32 threads (one wave per 16-row q-tile)
// ---------------------------------------------------------------------------
__global__ void k_argmax(const u16* __restrict__ wfs, const u16* __restrict__ wrt,
                         int* __restrict__ refidx, float* __restrict__ out_ridx, int q) {
    int qt = blockIdx.x, s = blockIdx.y, b = blockIdx.z;
    int lane = threadIdx.x;
    int ln15 = lane & 15;

    v16bf afr[4];
#pragma unroll
    for (int k4 = 0; k4 < 4; ++k4) {
        int qrow = qt * 16 + ln15;
        if (qrow >= q) qrow = q - 1;          // clamp (keeps EXEC all-ones)
        afr[k4] = load_a_frag(wfs + ((size_t)(b * q + qrow)) * CC, k4 * 32, lane);
    }

    float best[8];
    int   bidx[8];
#pragma unroll
    for (int r = 0; r < 8; ++r) { best[r] = -3.0e38f; bidx[r] = 0; }

    const u16* Bbase = wrt + (size_t)(b * NREF + s) * HW * CC;
    for (int ht = 0; ht < HW / 16; ++ht) {
        const u16* brow = Bbase + (size_t)(ht * 16 + ln15) * CC;
        if (ht + 1 < HW / 16)
            __builtin_prefetch(brow + 16 * CC, 0, 1);   // global_prefetch next tile
        v8f acc;
#pragma unroll
        for (int r = 0; r < 8; ++r) acc[r] = 0.0f;
#pragma unroll
        for (int k4 = 0; k4 < 4; ++k4) {
            v16bf bf = load_b_frag(brow, k4 * 32, lane);
            acc = wmma_bf16(afr[k4], bf, acc);
        }
        int hcol = ht * 16 + ln15;                       // N of this lane
#pragma unroll
        for (int r = 0; r < 8; ++r) {
            float v = acc[r];
            if (v > best[r]) { best[r] = v; bidx[r] = hcol; }
        }
    }

    // argmax butterfly across the 16-lane half (masks 1,2,4,8 stay in-half)
#pragma unroll
    for (int r = 0; r < 8; ++r) {
        float v = best[r];
        int   i = bidx[r];
        for (int m = 1; m <= 8; m <<= 1) {
            float ov = __shfl_xor(v, m, 32);
            int   oi = __shfl_xor(i, m, 32);
            if (ov > v || (ov == v && oi < i)) { v = ov; i = oi; }
        }
        if (ln15 == 0) {
            int qrow = qt * 16 + r + ((lane >= 16) ? 8 : 0);
            if (qrow < q) {
                size_t o = (size_t)(b * NREF + s) * q + qrow;
                refidx[o] = i;
                out_ridx[o] = (float)i;
            }
        }
    }
}

// ---------------------------------------------------------------------------
// Kernel 5: build fused LSTM input X0 (T,32,128) bf16 (matches reshape(b,-1,c))
// ---------------------------------------------------------------------------
__global__ void k_x0(const float* __restrict__ feat, const float* __restrict__ refs,
                     const int* __restrict__ idxs, const int* __restrict__ refidx,
                     const int* __restrict__ indexp, u16* __restrict__ X0,
                     int q, int T) {
    long long gid = (long long)blockIdx.x * blockDim.x + threadIdx.x;
    long long tot = (long long)BB * T * CC;
    if (gid >= tot) return;
    int b  = (int)(gid / ((long long)T * CC));
    int g  = (int)(gid % ((long long)T * CC));    // t*128 + j, flat within batch
    int t  = g / CC, j = g % CC;
    int cq = CC * q;
    int slot = g / cq;
    int rem  = g % cq;
    int ci = rem / q, qi = rem % q;
    float v;
    if (slot == 0) {
        v = feat[((size_t)(b * CC + ci)) * HW + idxs[b * q + qi]];
    } else {
        int s = slot - 1;
        int index = *indexp;
        int n0 = (s < index) ? s : s + 1;
        int rid = refidx[(size_t)(b * NREF + s) * q + qi];
        v = refs[(((size_t)(b * NT + n0) * CC) + ci) * HW + rid];
    }
    X0[((size_t)t * BB + b) * CC + j] = f2bf(v);
}

// ---------------------------------------------------------------------------
// Kernel 6: f32 -> bf16 weight convert (W_ih, all layers)
// ---------------------------------------------------------------------------
__global__ void k_wcvt(const float* __restrict__ W, u16* __restrict__ Wb, int n) {
    int i = blockIdx.x * blockDim.x + threadIdx.x;
    if (i < n) Wb[i] = f2bf(W[i]);
}

// ---------------------------------------------------------------------------
// Kernel 7: Gin = X @ W_ih^T + b_ih + b_hh, stored TRANSPOSED as (t, n, m32)
// so each timestep slice is one contiguous 32 KB block and each lane's 8
// batch values are contiguous (single b128 store here, b128 load in k_lstm).
// grid (2T, 32), 32 threads = one wave per 16x16 tile, WMMA bf16
// ---------------------------------------------------------------------------
__global__ void k_gin(const u16* __restrict__ X, const u16* __restrict__ Wb,
                      const float* __restrict__ bih, const float* __restrict__ bhh,
                      u16* __restrict__ Gin) {
    int mt = blockIdx.x, nt = blockIdx.y, lane = threadIdx.x;
    int ln15 = lane & 15;
    int n = nt * 16 + ln15;
    float bias = bih[n] + bhh[n];
    const u16* arow = X + ((size_t)(mt * 16 + ln15)) * CC;
    const u16* brow = Wb + ((size_t)n) * CC;
    v8f acc;
#pragma unroll
    for (int r = 0; r < 8; ++r) acc[r] = 0.0f;
#pragma unroll
    for (int k4 = 0; k4 < 4; ++k4)
        acc = wmma_bf16(load_a_frag(arow, k4 * 32, lane),
                        load_b_frag(brow, k4 * 32, lane), acc);
    int hi = (lane >= 16) ? 8 : 0;
    Pack8 pk;
#pragma unroll
    for (int r = 0; r < 8; ++r) pk.h[r] = f2bf(acc[r] + bias);
    // (t, n, m): t = mt/2, m = (mt&1)*16 + hi + r  (r contiguous)
    size_t o = ((size_t)(mt >> 1) * GID + n) * BB + (size_t)((mt & 1) * 16 + hi);
    *(v4u*)(Gin + o) = pk.q;
}

// ---------------------------------------------------------------------------
// Kernel 8: persistent LSTM recurrence, one workgroup (8 waves)
// W_hh bf16 B-fragments live in registers; h ping-pongs through 16 KB LDS.
// Next timestep's 32 KB Gin slice is prefetched into LDS by the Tensor Data
// Mover (tensor_load_to_lds + s_wait_tensorcnt), double-buffered.
// Wave w owns hidden columns j = 16w..16w+15 across all 4 gates.
// ---------------------------------------------------------------------------
__global__ void __launch_bounds__(256, 1)
k_lstm(const u16* __restrict__ Gin, const float* __restrict__ Whh,
       u16* __restrict__ Hout, float* __restrict__ finalOut, int T) {
    __shared__ __align__(16) u16 hbuf[2][BB * CC];
#if LSTM_TDM
    __shared__ __align__(16) u16 gbuf[2][STEPG];   // 2 x 32 KB double buffer
#endif

    int tid  = threadIdx.x;
    int lane = tid & 31;
    int w    = tid >> 5;
    int ln15 = lane & 15;
    int hi   = (lane >= 16) ? 8 : 0;
    int j    = 16 * w + ln15;

    // Preload W_hh B-fragments for this lane: n = g*128 + j, 4 k-steps each.
    FragU bfr[4][4];
#pragma unroll
    for (int g = 0; g < 4; ++g) {
        const float* wrow = Whh + ((size_t)(g * CC + j)) * CC;
#pragma unroll
        for (int k4 = 0; k4 < 4; ++k4) {
            const float* p = wrow + k4 * 32 + ((lane < 16) ? 0 : 16);
            FragU f;
#pragma unroll
            for (int e = 0; e < 16; ++e) f.h[e] = f2bf(p[e]);
            bfr[g][k4] = f;
        }
    }
    for (int i = tid; i < 2 * BB * CC; i += 256) ((u16*)hbuf)[i] = 0;
    float cst[2][8];
#pragma unroll
    for (int mt = 0; mt < 2; ++mt)
#pragma unroll
        for (int r = 0; r < 8; ++r) cst[mt][r] = 0.0f;

#if LSTM_TDM
    if (w == 0) {                       // wave-uniform: only wave 0 issues TDM
        tdm_load_gin(Gin, (u32)(size_t)(void*)&gbuf[0][0]);
        __builtin_amdgcn_s_wait_tensorcnt(0);
    }
#endif
    __syncthreads();

    for (int t = 0; t < T; ++t) {
        const u16* hb = hbuf[t & 1];
        u16*       hn = hbuf[(t + 1) & 1];

#if LSTM_TDM
        if (w == 0 && t + 1 < T)        // TDM prefetch next step's Gin slice
            tdm_load_gin(Gin + (size_t)(t + 1) * STEPG,
                         (u32)(size_t)(void*)&gbuf[(t + 1) & 1][0]);
        const u16* gsl = gbuf[t & 1];
#else
        const u16* gsl = Gin + (size_t)t * STEPG;
#endif

        v8f acc[2][4];
#pragma unroll
        for (int mt = 0; mt < 2; ++mt)
#pragma unroll
            for (int g = 0; g < 4; ++g)
#pragma unroll
                for (int r = 0; r < 8; ++r) acc[mt][g][r] = 0.0f;

#pragma unroll
        for (int k4 = 0; k4 < 4; ++k4) {
            v16bf a0 = load_a_frag(hb + (size_t)ln15 * CC,        k4 * 32, lane);
            v16bf a1 = load_a_frag(hb + (size_t)(16 + ln15) * CC, k4 * 32, lane);
#pragma unroll
            for (int g = 0; g < 4; ++g) {
                acc[0][g] = wmma_bf16(a0, bfr[g][k4].v, acc[0][g]);
                acc[1][g] = wmma_bf16(a1, bfr[g][k4].v, acc[1][g]);
            }
        }

#pragma unroll
        for (int mt = 0; mt < 2; ++mt) {
            // packed gate biases: (n, m) layout -> 8 contiguous batch values
            Pack8 pi, pf, pg, po;
            int mb = mt * 16 + hi;
            pi.q = *(const v4u*)(gsl + ((size_t)(0 * CC + j)) * BB + mb);
            pf.q = *(const v4u*)(gsl + ((size_t)(1 * CC + j)) * BB + mb);
            pg.q = *(const v4u*)(gsl + ((size_t)(2 * CC + j)) * BB + mb);
            po.q = *(const v4u*)(gsl + ((size_t)(3 * CC + j)) * BB + mb);
#pragma unroll
            for (int r = 0; r < 8; ++r) {
                int m = mb + r;                            // batch index
                float gi = acc[mt][0][r] + bf2f(pi.h[r]);
                float gf = acc[mt][1][r] + bf2f(pf.h[r]);
                float gg = acc[mt][2][r] + bf2f(pg.h[r]);
                float go = acc[mt][3][r] + bf2f(po.h[r]);
                float c  = sigm(gf) * cst[mt][r] + sigm(gi) * tanh_fast(gg);
                cst[mt][r] = c;
                float h  = sigm(go) * tanh_fast(c);
                u16 hb16 = f2bf(h);
                hn[m * CC + j] = hb16;
                Hout[((size_t)t * BB + m) * CC + j] = hb16;
                if (finalOut != nullptr && t == T - 1)
                    finalOut[m * CC + j] = h;
            }
        }

#if LSTM_TDM
        if (w == 0) __builtin_amdgcn_s_wait_tensorcnt(0);
#endif
        __syncthreads();
    }
}

// ---------------------------------------------------------------------------
// Host: kernel_launch
// ---------------------------------------------------------------------------
extern "C" void kernel_launch(void* const* d_in, const int* in_sizes, int n_in,
                              void* d_out, int out_size, void* d_ws, size_t ws_size,
                              hipStream_t stream) {
    const float* feat   = (const float*)d_in[0];
    const float* refs   = (const float*)d_in[1];
    const int*   indexp = (const int*)d_in[2];
    // d_in[3] = similarity (unused by reference output)
    const float* mask   = (const float*)d_in[4];
    const float* W_ih   = (const float*)d_in[5];
    const float* W_hh   = (const float*)d_in[6];
    const float* b_ih   = (const float*)d_in[7];
    const float* b_hh   = (const float*)d_in[8];
    float* out = (float*)d_out;

    // out = rnn(32*128) + indices(32*q) + ref_idx(32*8*q) => q from out_size
    int q = (out_size - BB * CC) / (BB * NT);
    if (q < 1) q = 1;
    int T  = NT * q;
    int qT = (q + 15) / 16;

    float* out_rnn  = out;
    float* out_idx  = out + BB * CC;
    float* out_ridx = out + BB * CC + BB * q;

    // workspace carve (256B aligned)
    char*  ws  = (char*)d_ws;
    size_t off = 0;
    auto carve = [&](size_t bytes) -> void* {
        void* p = (void*)(ws + off);
        off = (off + bytes + 255) & ~(size_t)255;
        return p;
    };
    int*  idxs   = (int*)carve((size_t)BB * q * 4);
    int*  tmp    = (int*)carve((size_t)BB * HW * 4);
    int*  refidx = (int*)carve((size_t)BB * NREF * q * 4);
    u16*  wfs    = (u16*)carve((size_t)BB * q * CC * 2);
    u16*  wrt    = (u16*)carve((size_t)BB * NREF * HW * CC * 2);
    u16*  X0     = (u16*)carve((size_t)T * BB * CC * 2);
    u16*  Wihb   = (u16*)carve((size_t)3 * GID * CC * 2);
    u16*  Gin    = (u16*)carve((size_t)T * STEPG * 2);
    u16*  HA     = (u16*)carve((size_t)T * BB * CC * 2);
    u16*  HB     = (u16*)carve((size_t)T * BB * CC * 2);
    (void)ws_size; (void)in_sizes; (void)n_in;

    // Phase 1: indices + normalizations
    k_indices<<<BB, 1, 0, stream>>>(mask, indexp, idxs, tmp, out_idx, q);
    k_wfs<<<dim3(q, BB), CC, 0, stream>>>(feat, idxs, wfs, q);
    k_wrt<<<BB * NREF * HW, CC, 0, stream>>>(refs, indexp, wrt);

    // Phase 2: WMMA similarity GEMM + fused argmax
    k_argmax<<<dim3(qT, NREF, BB), 32, 0, stream>>>(wfs, wrt, refidx, out_ridx, q);

    // Phase 3: fused LSTM input + weights
    {
        long long tot = (long long)BB * T * CC;
        int blocks = (int)((tot + 255) / 256);
        k_x0<<<blocks, 256, 0, stream>>>(feat, refs, idxs, refidx, indexp, X0, q, T);
    }
    {
        int n = 3 * GID * CC;
        k_wcvt<<<(n + 255) / 256, 256, 0, stream>>>(W_ih, Wihb, n);
    }

    // Phase 4: 3-layer LSTM: big WMMA Gin GEMM + persistent recurrent kernel
    for (int l = 0; l < 3; ++l) {
        const u16* X = (l == 0) ? X0 : ((l == 1) ? HA : HB);
        u16*       H = (l == 1) ? HB : HA;     // l0->HA, l1->HB, l2->HA
        k_gin<<<dim3(2 * T, GID / 16), 32, 0, stream>>>(
            X, Wihb + (size_t)l * GID * CC,
            b_ih + (size_t)l * GID, b_hh + (size_t)l * GID, Gin);
        k_lstm<<<1, 256, 0, stream>>>(
            Gin, W_hh + (size_t)l * GID * CC, H,
            (l == 2) ? out_rnn : nullptr, T);
    }
}
